// DynamicGraphClassification_34677565948080
// MI455X (gfx1250) — matrix operand order repaired and checked by reading, not compiled, not run on hardware
//
#include <hip/hip_runtime.h>
#include <hip/hip_bf16.h>

typedef _Float16 v16h __attribute__((ext_vector_type(16)));
typedef _Float16 v8h  __attribute__((ext_vector_type(8)));
typedef float    v8f  __attribute__((ext_vector_type(8)));
typedef unsigned int v4u __attribute__((ext_vector_type(4)));
typedef int      v8i  __attribute__((ext_vector_type(8)));
typedef int      v4i  __attribute__((ext_vector_type(4)));

#define NPTS   16384
#define KNB    20
#define BN_EPS 1e-5f
#define SLOPE  0.2f

// ---------------------------------------------------------------------------
// WMMA helpers (CDNA5 ISA 7.12.2 layouts, wave32)
// ---------------------------------------------------------------------------
__device__ __forceinline__ v8f wmma_f16(v16h a, v16h b, v8f c) {
  return __builtin_amdgcn_wmma_f32_16x16x32_f16(
      false, a, false, b, (short)0, c, false, false);
}

// A 16x32 f16 fragment from a row-major tile (element (m,k) at base[m*ld+k]).
// Per-lane data is one contiguous 32B run (k = 16h + s) -> 2 x b128 loads.
__device__ __forceinline__ v16h load_a_frag(const _Float16* base, int ld, int lane) {
  const int m   = lane & 15;
  const int hiK = (lane >> 4) << 4;
  const _Float16* row = base + m * ld + hiK;
  v8h lo = *(const v8h*)(row);
  v8h hi = *(const v8h*)(row + 8);
  return __builtin_shufflevector(lo, hi, 0, 1, 2, 3, 4, 5, 6, 7,
                                 8, 9, 10, 11, 12, 13, 14, 15);
}

// B 32x16 fragment where B[k][n] = Xj[n][k] (Gram tile). 32B contiguous/lane.
__device__ __forceinline__ v16h load_b_frag_colmajor(const _Float16* base, int ld, int lane) {
  const int n  = lane & 15;
  const int k0 = (lane >> 4) << 4;
  return *(const v16h*)(base + n * ld + k0);
}

// B fragment from pre-packed weights: [f][ct][lane][16 halves] contiguous.
__device__ __forceinline__ v16h load_b_frag_packed(const _Float16* Wp, int nct,
                                                   int f, int ct, int lane) {
  return *(const v16h*)(Wp + ((((size_t)f * nct + ct) << 5) + lane) * 16);
}

// ---------------------------------------------------------------------------
// Tensor Data Mover: 2D tile (f16) global -> LDS.  D# per CDNA5 ISA ch.8.
// ---------------------------------------------------------------------------
__device__ __forceinline__ void tdm_load_tile_2d_f16(unsigned lds_off, const void* gptr,
                                                     unsigned elems_per_row,
                                                     unsigned rows) {
  const unsigned long long ga = (unsigned long long)gptr;
  v4u g0 = {};
  g0[0] = 1u;                                            // count=1 (user mode)
  g0[1] = lds_off;                                       // lds_addr
  g0[2] = (unsigned)ga;                                  // global_addr[31:0]
  g0[3] = (unsigned)((ga >> 32) & 0x01FFFFFFull) | (2u << 30);  // addr[56:32] | type=2
  v8i g1 = {};
  g1[0] = (int)(1u << 16);                               // data_size = 1 (2 bytes)
  g1[1] = (int)((elems_per_row & 0xFFFFu) << 16);        // tensor_dim0[15:0]
  g1[2] = (int)(((elems_per_row >> 16) & 0xFFFFu) |
                ((rows & 0xFFFFu) << 16));               // tdim0 hi | tensor_dim1 lo
  g1[3] = (int)(((rows >> 16) & 0xFFFFu) |
                ((elems_per_row & 0xFFFFu) << 16));      // tdim1 hi | tile_dim0
  g1[4] = (int)(rows & 0xFFFFu);                         // tile_dim1 (tile_dim2 = 0)
  g1[5] = (int)elems_per_row;                            // tensor_dim0_stride lo32
  g1[6] = 0;
  g1[7] = 0;
  v4i g2 = {};
  v4i g3 = {};
#if __clang_major__ >= 23
  v8i g4 = {};
  __builtin_amdgcn_tensor_load_to_lds(g0, g1, g2, g3, g4, 0);
#else
  __builtin_amdgcn_tensor_load_to_lds(g0, g1, g2, g3, 0);
#endif
}

// ---------------------------------------------------------------------------
// 1) feature prep: f32 [N,F] (strided) -> zero-padded f16 [N,FPAD] + sq norms
// ---------------------------------------------------------------------------
template <int FPAD>
__global__ __launch_bounds__(128) void prep_feat_kernel(
    const float* __restrict__ x, int xstride, int F,
    _Float16* __restrict__ xh, float* __restrict__ sq) {
  __shared__ float red[128];
  const int i = blockIdx.x;
  const int t = threadIdx.x;
  float v = 0.f;
  if (t < FPAD) {
    v = (t < F) ? x[(size_t)i * xstride + t] : 0.f;
    xh[(size_t)i * FPAD + t] = (_Float16)v;
  }
  red[t] = v * v;
  __syncthreads();
#pragma unroll
  for (int s = 64; s > 0; s >>= 1) {
    if (t < s) red[t] += red[t + s];
    __syncthreads();
  }
  if (t == 0) sq[i] = red[0];
}

// ---------------------------------------------------------------------------
// 2) fused Gram + top-K kNN. Block = 16 query rows, 8 waves.
//    TDM double-buffers 128-column f16 chunks into LDS (wave 0 issues the DMA
//    for chunk ch+1 while all waves WMMA on chunk ch); per-wave top-20 lists
//    merged at the end.
// ---------------------------------------------------------------------------
template <int FPAD>
__global__ __launch_bounds__(256) void knn_kernel(
    const _Float16* __restrict__ xh, const float* __restrict__ sq,
    int* __restrict__ knn_idx, int N) {
  extern __shared__ unsigned char smem_raw[];
  _Float16* Xj  = (_Float16*)smem_raw;                       // 2 x 128*FPAD (double buf)
  float* distT  = (float*)(smem_raw + (size_t)2 * 128 * FPAD * sizeof(_Float16));
  float* bestd  = distT + 8 * 256;                           // 8*16*KNB
  int*   besti  = (int*)(bestd + 8 * 16 * KNB);              // 8*16*KNB

  const int t    = threadIdx.x;
  const int wave = t >> 5;
  const int lane = t & 31;
  const int i0   = blockIdx.x * 16;
  const int nchunks = N / 128;

  // kick off DMA of chunk 0 into buffer 0
  if (wave == 0) {
    tdm_load_tile_2d_f16((unsigned)(unsigned long long)(const void*)Xj,
                         xh, FPAD, 128);
  }

  for (int p = t; p < 8 * 16 * KNB; p += 256) { bestd[p] = 3.0e38f; besti[p] = 0; }

  // preload A fragments for the 16 query rows (vector gathers from global f16)
  v16h afr[FPAD / 32];
#pragma unroll
  for (int f = 0; f < FPAD / 32; ++f)
    afr[f] = load_a_frag(xh + (size_t)i0 * FPAD + f * 32, FPAD, lane);

  float sqi_r[8];
#pragma unroll
  for (int r = 0; r < 8; ++r) sqi_r[r] = sq[i0 + ((lane >> 4) << 3) + r];

  for (int ch = 0; ch < nchunks; ++ch) {
    const int c0 = ch * 128;
    _Float16* buf = Xj + (size_t)(ch & 1) * 128 * FPAD;
    if (wave == 0) {
      if (ch + 1 < nchunks) {  // issue next chunk, then wait for current one
        _Float16* nbuf = Xj + (size_t)((ch + 1) & 1) * 128 * FPAD;
        tdm_load_tile_2d_f16((unsigned)(unsigned long long)(const void*)nbuf,
                             xh + (size_t)(c0 + 128) * FPAD, FPAD, 128);
        __builtin_amdgcn_s_wait_tensorcnt(1);
      } else {
        __builtin_amdgcn_s_wait_tensorcnt(0);
      }
    }
    __syncthreads();  // chunk ch visible to all waves

    // wave computes dot tile vs columns [c0 + wave*16, +16)
    v8f acc = {};
#pragma unroll
    for (int f = 0; f < FPAD / 32; ++f) {
      v16h b = load_b_frag_colmajor(buf + (wave * 16) * FPAD + f * 32, FPAD, lane);
      acc = wmma_f16(afr[f], b, acc);
    }
    float* dt = distT + wave * 256;
    const int nl  = lane & 15;
    const float sqc = sq[c0 + wave * 16 + nl];
#pragma unroll
    for (int r = 0; r < 8; ++r) {
      const int M = ((lane >> 4) << 3) + r;
      dt[M * 16 + nl] = sqi_r[r] + sqc - 2.0f * acc[r];
    }
    // per-wave top-K: one lane per query row, threshold + insertion sort
    if (lane < 16) {
      float* bd = bestd + (wave * 16 + lane) * KNB;
      int*   bi = besti + (wave * 16 + lane) * KNB;
      const float* drow = dt + lane * 16;
      for (int c = 0; c < 16; ++c) {
        const float d = drow[c];
        if (d < bd[KNB - 1]) {
          int p = KNB - 1;
          while (p > 0 && bd[p - 1] > d) {
            bd[p] = bd[p - 1]; bi[p] = bi[p - 1]; --p;
          }
          bd[p] = d; bi[p] = c0 + wave * 16 + c;
        }
      }
    }
    __syncthreads();  // all waves done with buf before it is overwritten
  }

  // merge the 8 per-wave lists for each of the 16 rows
  if (t < 16) {
    float md[KNB]; int mi[KNB];
#pragma unroll
    for (int e = 0; e < KNB; ++e) { md[e] = 3.0e38f; mi[e] = 0; }
    for (int w = 0; w < 8; ++w) {
      const float* bd = bestd + (w * 16 + t) * KNB;
      const int*   bi = besti + (w * 16 + t) * KNB;
      for (int e = 0; e < KNB; ++e) {
        const float d = bd[e];
        if (d < md[KNB - 1]) {
          int p = KNB - 1;
          while (p > 0 && md[p - 1] > d) { md[p] = md[p - 1]; mi[p] = mi[p - 1]; --p; }
          md[p] = d; mi[p] = bi[e];
        }
      }
    }
    for (int e = 0; e < KNB; ++e) knn_idx[(size_t)(i0 + t) * KNB + e] = mi[e];
  }
}

// ---------------------------------------------------------------------------
// 3) EdgeConv: block = 4 nodes = 80 edges. Build [80, 2*FPAD] f16 edge
//    features in LDS, WMMA against pre-packed f16 weights, fused BN+LeakyReLU
//    epilogue, mean over K=20 neighbors.
// ---------------------------------------------------------------------------
template <int FPAD>
__global__ __launch_bounds__(256) void edge_conv_kernel(
    const float* __restrict__ x, int xstride, int F,
    const int* __restrict__ knn_idx,
    const _Float16* __restrict__ Wp,   // packed [KD/32][4][32][16] f16
    const float* __restrict__ g, const float* __restrict__ bb,
    const float* __restrict__ mm, const float* __restrict__ vv,
    float* __restrict__ out, int ostride) {
  constexpr int KD = 2 * FPAD;
  constexpr int NODES = 4;
  constexpr int EDGES = NODES * KNB;  // 80
  extern __shared__ unsigned char smem_raw[];
  _Float16* Ash = (_Float16*)smem_raw;              // EDGES x KD
  float* hbuf   = (float*)(Ash + EDGES * KD);       // EDGES x 64
  int* nbr      = (int*)(hbuf + EDGES * 64);        // EDGES

  const int t    = threadIdx.x;
  const int wave = t >> 5;
  const int lane = t & 31;
  const int i0   = blockIdx.x * NODES;

  if (t < EDGES) nbr[t] = knn_idx[(size_t)(i0 + t / KNB) * KNB + (t % KNB)];
  __syncthreads();

  // stage edge features e = [x_i, x_j - x_i] (zero-padded) as f16
  for (int p = t; p < EDGES * FPAD; p += 256) {
    const int e  = p / FPAD;
    const int kk = p - e * FPAD;
    const int i  = i0 + e / KNB;
    const int j  = nbr[e];
    float xi = 0.f, xj = 0.f;
    if (kk < F) {
      xi = x[(size_t)i * xstride + kk];
      xj = x[(size_t)j * xstride + kk];
    }
    Ash[e * KD + kk]        = (_Float16)xi;
    Ash[e * KD + FPAD + kk] = (_Float16)(xj - xi);
  }
  __syncthreads();

  // (EDGES/16) edge-tiles x 4 col-tiles of 16 outputs = 20 WMMA jobs / 8 waves
  for (int job = wave; job < (EDGES / 16) * 4; job += 8) {
    const int et = job >> 2;
    const int ct = job & 3;
    v8f acc = {};
#pragma unroll
    for (int f = 0; f < KD / 32; ++f) {
      v16h a = load_a_frag(Ash + (et * 16) * KD + f * 32, KD, lane);
      v16h b = load_b_frag_packed(Wp, 4, f, ct, lane);
      acc = wmma_f16(a, b, acc);
    }
    const int nl = lane & 15;
    const int o  = ct * 16 + nl;
    const float gg = g[o], bo = bb[o], mo = mm[o], inv = rsqrtf(vv[o] + BN_EPS);
#pragma unroll
    for (int r = 0; r < 8; ++r) {
      const int e = et * 16 + ((lane >> 4) << 3) + r;
      float v = (acc[r] - mo) * inv * gg + bo;
      v = (v >= 0.f) ? v : SLOPE * v;
      hbuf[e * 64 + o] = v;
    }
  }
  __syncthreads();

  {  // mean aggregation over the 20 neighbors of each node
    const int node = t >> 6;
    const int o    = t & 63;
    float s = 0.f;
#pragma unroll
    for (int k = 0; k < KNB; ++k) s += hbuf[(node * KNB + k) * 64 + o];
    out[(size_t)(i0 + node) * ostride + o] = s * (1.0f / KNB);
  }
}

// ---------------------------------------------------------------------------
// 4) Head GEMM: [N, Kd] x [Kd, Nout] (packed weights), optional bias/BN/act.
// ---------------------------------------------------------------------------
__global__ __launch_bounds__(256) void gemm_head_kernel(
    const float* __restrict__ X, int Kd,
    const _Float16* __restrict__ Wp, int Nout,
    const float* __restrict__ bias,
    const float* __restrict__ g, const float* __restrict__ bb,
    const float* __restrict__ mm, const float* __restrict__ vv,
    int do_act, float* __restrict__ out) {
  extern __shared__ unsigned char smem_raw[];
  _Float16* Xs = (_Float16*)smem_raw;  // 16 x Kd
  const int t    = threadIdx.x;
  const int wave = t >> 5;
  const int lane = t & 31;
  const int i0   = blockIdx.x * 16;

  __builtin_prefetch(Wp + (size_t)t * 64, 0, 2);
  for (int p = t; p < 16 * Kd; p += 256)
    Xs[p] = (_Float16)X[(size_t)(i0 + p / Kd) * Kd + (p % Kd)];
  __syncthreads();

  const int nct = Nout / 16;
  for (int ct = wave; ct < nct; ct += 8) {
    v8f acc = {};
    for (int f = 0; f < Kd / 32; ++f) {
      v16h a = load_a_frag(Xs + f * 32, Kd, lane);
      v16h b = load_b_frag_packed(Wp, nct, f, ct, lane);
      acc = wmma_f16(a, b, acc);
    }
    const int nl = lane & 15;
    const int o  = ct * 16 + nl;
    const float add = bias ? bias[o] : 0.f;
    float gg = 1.f, bo = 0.f, mo = 0.f, inv = 1.f;
    if (g) { gg = g[o]; bo = bb[o]; mo = mm[o]; inv = rsqrtf(vv[o] + BN_EPS); }
#pragma unroll
    for (int r = 0; r < 8; ++r) {
      const int row = i0 + ((lane >> 4) << 3) + r;
      float v = acc[r] + add;
      if (g) v = (v - mo) * inv * gg + bo;
      if (do_act) v = (v >= 0.f) ? v : SLOPE * v;
      out[(size_t)row * Nout + o] = v;
    }
  }
}

// 5) tiny logits head: [N,128] x [128,2] + b
__global__ void logits_kernel(const float* __restrict__ feat,
                              const float* __restrict__ Wclf,
                              const float* __restrict__ bclf,
                              float* __restrict__ logits, int N) {
  const int p = blockIdx.x * blockDim.x + threadIdx.x;
  if (p >= N * 2) return;
  const int i = p >> 1, o = p & 1;
  const float* f = feat + (size_t)i * 128;
  float s = bclf[o];
#pragma unroll 4
  for (int c = 0; c < 128; ++c) s += f[c] * Wclf[c * 2 + o];
  logits[p] = s;
}

// ---------------------------------------------------------------------------
// Weight conversion into WMMA-B fragment-major packing:
//   Wp[((f*nct + ct)*32 + lane)*16 + h] = W[(f*32 + (lane>>4)*16 + h)][ct*16 + (lane&15)]
// split=1 applies the EdgeConv zero-split padding ([F|pad] twice along K).
// ---------------------------------------------------------------------------
__global__ void cvt_w_packed_kernel(const float* __restrict__ W, _Float16* __restrict__ Wp,
                                    int Kd, int Nout, int F, int Fpad, int split) {
  const int p = blockIdx.x * blockDim.x + threadIdx.x;
  if (p >= Kd * Nout) return;
  const int nct  = Nout >> 4;
  const int h    = p & 15;
  const int lane = (p >> 4) & 31;
  const int rem  = p >> 9;
  const int ct   = rem % nct;
  const int f    = rem / nct;
  const int kk   = f * 32 + ((lane >> 4) << 4) + h;
  const int n    = ct * 16 + (lane & 15);
  int src;
  if (split) {
    src = (kk < Fpad) ? (kk < F ? kk : -1)
                      : ((kk - Fpad) < F ? F + (kk - Fpad) : -1);
  } else {
    src = kk;
  }
  Wp[p] = (src >= 0) ? (_Float16)W[(size_t)src * Nout + n] : (_Float16)0.0f;
}

// ---------------------------------------------------------------------------
extern "C" void kernel_launch(void* const* d_in, const int* in_sizes, int n_in,
                              void* d_out, int out_size, void* d_ws, size_t ws_size,
                              hipStream_t stream) {
  (void)in_sizes; (void)n_in; (void)out_size; (void)ws_size;
  const int N = NPTS, F0 = 116;

  const float* x    = (const float*)d_in[0];
  const float* W1   = (const float*)d_in[1];
  const float *g1 = (const float*)d_in[2],  *b1 = (const float*)d_in[3],
              *m1 = (const float*)d_in[4],  *v1 = (const float*)d_in[5];
  const float* W2   = (const float*)d_in[6];
  const float *g2 = (const float*)d_in[7],  *b2 = (const float*)d_in[8],
              *m2 = (const float*)d_in[9],  *v2 = (const float*)d_in[10];
  const float* W3   = (const float*)d_in[11];
  const float *g3 = (const float*)d_in[12], *b3 = (const float*)d_in[13],
              *m3 = (const float*)d_in[14], *v3 = (const float*)d_in[15];
  const float* Wc   = (const float*)d_in[16];
  const float* Wm1  = (const float*)d_in[17];
  const float* bm1  = (const float*)d_in[18];
  const float *g4 = (const float*)d_in[19], *b4 = (const float*)d_in[20],
              *m4 = (const float*)d_in[21], *v4 = (const float*)d_in[22];
  const float* Wm2  = (const float*)d_in[23];
  const float* bm2  = (const float*)d_in[24];
  const float *g5 = (const float*)d_in[25], *b5 = (const float*)d_in[26],
              *m5 = (const float*)d_in[27], *v5 = (const float*)d_in[28];
  const float* Wclf = (const float*)d_in[29];
  const float* bclf = (const float*)d_in[30];

  // ---- workspace carve
  unsigned char* ws = (unsigned char*)d_ws;
  auto alloc = [&](size_t bytes) {
    unsigned char* p = ws;
    ws += (bytes + 255) & ~(size_t)255;
    return p;
  };
  _Float16* xh   = (_Float16*)alloc((size_t)N * 128 * 2);
  float* sq      = (float*)alloc((size_t)N * 4);
  int* knn_idx   = (int*)alloc((size_t)N * KNB * 4);
  float* xcat    = (float*)alloc((size_t)N * 192 * 4);   // [x1 | x2 | x3]
  float* h512    = (float*)alloc((size_t)N * 512 * 4);
  float* h256    = (float*)alloc((size_t)N * 256 * 4);
  _Float16* W1p  = (_Float16*)alloc(256 * 64 * 2);
  _Float16* W2p  = (_Float16*)alloc(128 * 64 * 2);
  _Float16* W3p  = (_Float16*)alloc(128 * 64 * 2);
  _Float16* Wcp  = (_Float16*)alloc((size_t)192 * 512 * 2);
  _Float16* Wm1p = (_Float16*)alloc((size_t)512 * 256 * 2);
  _Float16* Wm2p = (_Float16*)alloc((size_t)256 * 128 * 2);

  float* logits  = (float*)d_out;                 // [N,2]
  float* feature = (float*)d_out + (size_t)N * 2; // [N,128]

  // ---- weight packing (f32 -> f16 fragment-major)
  cvt_w_packed_kernel<<<(256 * 64 + 255) / 256, 256, 0, stream>>>(W1, W1p, 256, 64, F0, 128, 1);
  cvt_w_packed_kernel<<<(128 * 64 + 255) / 256, 256, 0, stream>>>(W2, W2p, 128, 64, 64, 64, 1);
  cvt_w_packed_kernel<<<(128 * 64 + 255) / 256, 256, 0, stream>>>(W3, W3p, 128, 64, 64, 64, 1);
  cvt_w_packed_kernel<<<(192 * 512 + 255) / 256, 256, 0, stream>>>(Wc, Wcp, 192, 512, 0, 0, 0);
  cvt_w_packed_kernel<<<(512 * 256 + 255) / 256, 256, 0, stream>>>(Wm1, Wm1p, 512, 256, 0, 0, 0);
  cvt_w_packed_kernel<<<(256 * 128 + 255) / 256, 256, 0, stream>>>(Wm2, Wm2p, 256, 128, 0, 0, 0);

  auto knn_sh = [](int fpad) {
    return (size_t)2 * 128 * fpad * 2 + 8 * 256 * 4 + 2 * (8 * 16 * KNB * 4);
  };
  auto edge_sh = [](int kd) {
    return (size_t)80 * kd * 2 + (size_t)80 * 64 * 4 + 80 * 4;
  };

  // ---- layer 1 (F=116, pad 128)
  prep_feat_kernel<128><<<N, 128, 0, stream>>>(x, F0, F0, xh, sq);
  knn_kernel<128><<<N / 16, 256, knn_sh(128), stream>>>(xh, sq, knn_idx, N);
  edge_conv_kernel<128><<<N / 4, 256, edge_sh(256), stream>>>(
      x, F0, F0, knn_idx, W1p, g1, b1, m1, v1, xcat + 0, 192);

  // ---- layer 2 (input x1 = xcat[:, :64])
  prep_feat_kernel<64><<<N, 128, 0, stream>>>(xcat, 192, 64, xh, sq);
  knn_kernel<64><<<N / 16, 256, knn_sh(64), stream>>>(xh, sq, knn_idx, N);
  edge_conv_kernel<64><<<N / 4, 256, edge_sh(128), stream>>>(
      xcat, 192, 64, knn_idx, W2p, g2, b2, m2, v2, xcat + 64, 192);

  // ---- layer 3 (input x2 = xcat[:, 64:128])
  prep_feat_kernel<64><<<N, 128, 0, stream>>>(xcat + 64, 192, 64, xh, sq);
  knn_kernel<64><<<N / 16, 256, knn_sh(64), stream>>>(xh, sq, knn_idx, N);
  edge_conv_kernel<64><<<N / 4, 256, edge_sh(128), stream>>>(
      xcat + 64, 192, 64, knn_idx, W3p, g3, b3, m3, v3, xcat + 128, 192);

  // ---- head
  gemm_head_kernel<<<N / 16, 256, (size_t)16 * 192 * 2, stream>>>(
      xcat, 192, Wcp, 512, nullptr, nullptr, nullptr, nullptr, nullptr, 0, h512);
  gemm_head_kernel<<<N / 16, 256, (size_t)16 * 512 * 2, stream>>>(
      h512, 512, Wm1p, 256, bm1, g4, b4, m4, v4, 1, h256);
  gemm_head_kernel<<<N / 16, 256, (size_t)16 * 256 * 2, stream>>>(
      h256, 256, Wm2p, 128, bm2, g5, b5, m5, v5, 1, feature);
  logits_kernel<<<(N * 2 + 255) / 256, 256, 0, stream>>>(feature, Wclf, bclf, logits, N);
}
